// EGNNLayer_44959717655304
// MI455X (gfx1250) — matrix operand order repaired and checked by reading, not compiled, not run on hardware
//
#include <hip/hip_runtime.h>

typedef __attribute__((ext_vector_type(16))) _Float16 v16h;
typedef __attribute__((ext_vector_type(8)))  _Float16 v8h;
typedef __attribute__((ext_vector_type(8)))  float    v8f;

#define WAVES_PER_BLOCK 8
#define TPB 256
#define EDGE_TPW 4   // edge tiles (16 edges) per wave
#define NODE_TPW 2   // node tiles per wave
#define TSTR 72      // LDS tile row stride in halfs (64 + 8 pad: bank-conflict-free)

__device__ __forceinline__ float silu_f(float x) {
    return x * (1.0f / (1.0f + __expf(-x)));
}

__device__ __forceinline__ v8f wmma_f16(v16h a, v16h b, v8f c) {
    // (neg_a, A, neg_b, B, c_mod, C, reuse_a, reuse_b)
    return __builtin_amdgcn_wmma_f32_16x16x32_f16(false, a, false, b, (short)0, c, false, false);
}

// Transposed-weight A fragments, swizzled fragment-major in LDS:
//   lw[(((kt*4 + mt)*32 + lane)*16) + j] = W[K*64 + m]   (A row m = out-feature)
//   K = kt*32 + j + (lane>=16 ? 16:0),  m = mt*16 + (lane&15)
//   K == Kreal selects the bias row (folded bias), K > Kreal -> 0
__device__ __forceinline__ void fill_wswzA(_Float16* lw, const float* W, const float* bias,
                                           int KT, int Kreal, int tid) {
    const int total = KT * 4 * 32 * 16;
    for (int i = tid; i < total; i += TPB) {
        int j    = i & 15;
        int lane = (i >> 4) & 31;
        int mt   = (i >> 9) & 3;
        int kt   = i >> 11;
        int K = kt * 32 + j + ((lane & 16) ? 16 : 0);
        int m = mt * 16 + (lane & 15);
        float v = 0.0f;
        if (K < Kreal) v = W[K * 64 + m];
        else if (bias != nullptr && K == Kreal) v = bias[m];
        lw[i] = (_Float16)v;
    }
}

// A fragment (weights): one contiguous 32-byte per-lane block
__device__ __forceinline__ v16h ldA(const _Float16* lw, int kt, int mt, int lane) {
    return *(const v16h*)(lw + ((((kt << 2) + mt) << 5) + lane) * 16);
}

// B fragment (activations): 16 contiguous halfs from this lane's row
__device__ __forceinline__ v16h ldrow16(const _Float16* p) {
    const v8h a0 = *(const v8h*)p;
    const v8h a1 = *(const v8h*)(p + 8);
    return __builtin_shufflevector(a0, a1, 0, 1, 2, 3, 4, 5, 6, 7,
                                           8, 9, 10, 11, 12, 13, 14, 15);
}

// ---------------------------------------------------------------------------
// Kernel 0: zero h_agg/x_agg and convert feat -> f16
// ---------------------------------------------------------------------------
__global__ void egnn_prep(const float* __restrict__ feat, _Float16* __restrict__ feat16,
                          float* __restrict__ zbuf, long nz, long nf) {
    long i = (long)blockIdx.x * blockDim.x + threadIdx.x;
    if (i < nz) zbuf[i] = 0.0f;
    if (i < nf) feat16[i] = (_Float16)feat[i];
}

// ---------------------------------------------------------------------------
// Kernel 1: edge pipeline, transposed dataflow (A = W^T, B = activations,
// N = edge = lane). One wave handles EDGE_TPW tiles of 16 edges.
// ---------------------------------------------------------------------------
__global__ __launch_bounds__(TPB) void egnn_edge(
    const _Float16* __restrict__ feat16, const float* __restrict__ coord,
    const int* __restrict__ src, const int* __restrict__ dst,
    const float* __restrict__ We1, const float* __restrict__ be1,
    const float* __restrict__ We2, const float* __restrict__ be2,
    const float* __restrict__ Wc1, const float* __restrict__ bc1,
    const float* __restrict__ Wc2, const float* __restrict__ bc2,
    float* __restrict__ h_agg, float* __restrict__ x_agg, int E)
{
    __shared__ alignas(32) _Float16 lWe1[5 * 4 * 512];   // K padded to 160, row129=bias
    __shared__ alignas(32) _Float16 lWe2[2 * 4 * 512];
    __shared__ alignas(32) _Float16 lWc1[2 * 4 * 512];
    __shared__ alignas(32) _Float16 ltile[WAVES_PER_BLOCK][16 * TSTR];
    __shared__ alignas(32) float lWc2[64];
    __shared__ alignas(32) float lbe2[64];
    __shared__ alignas(32) float lbc1[64];
    __shared__ float lbc2s[1];

    const int tid = threadIdx.x;
    fill_wswzA(lWe1, We1, be1, 5, 129, tid);   // bias folded at K=129
    fill_wswzA(lWe2, We2, nullptr, 2, 64, tid);
    fill_wswzA(lWc1, Wc1, nullptr, 2, 64, tid);
    if (tid < 64) {
        lWc2[tid] = Wc2[tid];
        lbe2[tid] = be2[tid];
        lbc1[tid] = bc1[tid];
    }
    if (tid == 0) lbc2s[0] = bc2[0];
    __syncthreads();

    const int wave = tid >> 5;
    const int lane = tid & 31;
    const int ln   = lane & 15;   // this lane's edge within the tile (= B column)
    const int hi   = lane >> 4;
    const float bc2v = lbc2s[0];
    _Float16* trow = &ltile[wave][ln * TSTR];

    for (int t0 = 0; t0 < EDGE_TPW; ++t0) {
        const long tile = (long)blockIdx.x * (WAVES_PER_BLOCK * EDGE_TPW)
                        + wave * EDGE_TPW + t0;
        const long base = tile * 16;
        if (base >= E) break;

        long e = base + ln;
        const bool ev = (e < E);
        if (!ev) e = (long)E - 1;               // clamp; garbage cols never scattered
        const int si = src[e];
        const int di = dst[e];
        const float dx0 = coord[(long)si * 3 + 0] - coord[(long)di * 3 + 0];
        const float dx1 = coord[(long)si * 3 + 1] - coord[(long)di * 3 + 1];
        const float dx2 = coord[(long)si * 3 + 2] - coord[(long)di * 3 + 2];
        const float d2  = dx0 * dx0 + dx1 * dx1 + dx2 * dx2;
        const _Float16* fs = feat16 + (long)si * 64;
        const _Float16* fd = feat16 + (long)di * 64;

        // ---- layer 1: h1^T = silu(We1^T @ e_in^T), bias folded at K=129 ----
        v8f acc1[4] = {};
        #pragma unroll
        for (int kt = 0; kt < 4; ++kt) {
            const _Float16* sp = (kt < 2) ? fs : fd;
            v16h b = ldrow16(sp + (kt & 1) * 32 + hi * 16);
            #pragma unroll
            for (int mt = 0; mt < 4; ++mt)
                acc1[mt] = wmma_f16(ldA(lWe1, kt, mt, lane), b, acc1[mt]);
        }
        {   // kt = 4: K=128 -> d2, K=129 -> 1.0 (bias row), rest zero
            v16h b = {};
            if (!hi) { b[0] = (_Float16)d2; b[1] = (_Float16)1.0f; }
            #pragma unroll
            for (int mt = 0; mt < 4; ++mt)
                acc1[mt] = wmma_f16(ldA(lWe1, 4, mt, lane), b, acc1[mt]);
        }
        // silu -> ltile[edge][feature] (per-lane contiguous: vector stores)
        #pragma unroll
        for (int mt = 0; mt < 4; ++mt) {
            v8h hv;
            #pragma unroll
            for (int r = 0; r < 8; ++r) hv[r] = (_Float16)silu_f(acc1[mt][r]);
            *(v8h*)(trow + mt * 16 + hi * 8) = hv;
        }

        // ---- layer 2: h_e^T = silu(We2^T @ h1^T + be2) ----
        v8f acc2[4] = {};
        #pragma unroll
        for (int kt = 0; kt < 2; ++kt) {
            v16h b = ldrow16(trow + kt * 32 + hi * 16);
            #pragma unroll
            for (int mt = 0; mt < 4; ++mt)
                acc2[mt] = wmma_f16(ldA(lWe2, kt, mt, lane), b, acc2[mt]);
        }
        #pragma unroll
        for (int mt = 0; mt < 4; ++mt) {
            acc2[mt] += *(const v8f*)&lbe2[mt * 16 + hi * 8];
            v8h hv;
            #pragma unroll
            for (int r = 0; r < 8; ++r) {
                acc2[mt][r] = silu_f(acc2[mt][r]);
                hv[r] = (_Float16)acc2[mt][r];
            }
            *(v8h*)(trow + mt * 16 + hi * 8) = hv;
        }

        // ---- layer 3: c1^T = silu(Wc1^T @ h_e^T + bc1) ----
        v8f acc3[4] = {};
        #pragma unroll
        for (int kt = 0; kt < 2; ++kt) {
            v16h b = ldrow16(trow + kt * 32 + hi * 16);
            #pragma unroll
            for (int mt = 0; mt < 4; ++mt)
                acc3[mt] = wmma_f16(ldA(lWc1, kt, mt, lane), b, acc3[mt]);
        }

        // ---- coef = c1 . Wc2 + bc2 : per-lane partial over own features ----
        float p = 0.0f;
        #pragma unroll
        for (int mt = 0; mt < 4; ++mt) {
            v8f w = *(const v8f*)&lWc2[mt * 16 + hi * 8];
            v8f bb = *(const v8f*)&lbc1[mt * 16 + hi * 8];
            #pragma unroll
            for (int r = 0; r < 8; ++r)
                p += silu_f(acc3[mt][r] + bb[r]) * w[r];
        }
        // features split across the two lane-halves: one xor-add completes the dot
        p += __shfl_xor(p, 16, 32);
        const float cf = p + bc2v;

        // ---- scatter h_agg += h_e (lane owns its edge: no shuffles) ----
        if (ev) {
            float* hdst = h_agg + (long)di * 64 + hi * 8;
            #pragma unroll
            for (int mt = 0; mt < 4; ++mt)
                #pragma unroll
                for (int r = 0; r < 8; ++r)
                    atomicAdd(hdst + mt * 16 + r, acc2[mt][r]);
            // ---- scatter x_agg += dx * coef (low half only: one per edge) ----
            if (!hi) {
                atomicAdd(&x_agg[(long)di * 3 + 0], dx0 * cf);
                atomicAdd(&x_agg[(long)di * 3 + 1], dx1 * cf);
                atomicAdd(&x_agg[(long)di * 3 + 2], dx2 * cf);
            }
        }
    }
}

// ---------------------------------------------------------------------------
// Kernel 2: node pipeline, transposed dataflow. One wave: NODE_TPW tiles of 16.
// ---------------------------------------------------------------------------
__global__ __launch_bounds__(TPB) void egnn_node(
    const _Float16* __restrict__ feat16, const float* __restrict__ coord,
    const float* __restrict__ Wn1, const float* __restrict__ bn1,
    const float* __restrict__ Wn2, const float* __restrict__ bn2,
    const float* __restrict__ h_agg, const float* __restrict__ x_agg,
    float* __restrict__ out_h, float* __restrict__ out_x, int Nn)
{
    __shared__ alignas(32) _Float16 lWn1[4 * 4 * 512];
    __shared__ alignas(32) _Float16 lWn2[2 * 4 * 512];
    __shared__ alignas(32) _Float16 ltile[WAVES_PER_BLOCK][16 * TSTR];
    __shared__ alignas(32) float lbn1[64];
    __shared__ alignas(32) float lbn2[64];

    const int tid = threadIdx.x;
    fill_wswzA(lWn1, Wn1, nullptr, 4, 128, tid);
    fill_wswzA(lWn2, Wn2, nullptr, 2, 64, tid);
    if (tid < 64) { lbn1[tid] = bn1[tid]; lbn2[tid] = bn2[tid]; }
    __syncthreads();

    const int wave = tid >> 5;
    const int lane = tid & 31;
    const int ln   = lane & 15;
    const int hi   = lane >> 4;
    _Float16* trow = &ltile[wave][ln * TSTR];

    for (int t0 = 0; t0 < NODE_TPW; ++t0) {
        const long tile = (long)blockIdx.x * (WAVES_PER_BLOCK * NODE_TPW)
                        + wave * NODE_TPW + t0;
        const long base = tile * 16;
        if (base >= Nn) break;

        long nd = base + ln;
        const bool nv = (nd < Nn);
        if (!nv) nd = (long)Nn - 1;
        const _Float16* fp = feat16 + nd * 64;
        const float*    hp = h_agg  + nd * 64;

        // ---- layer 1: silu(Wn1^T @ [feat|h_agg]^T + bn1), K=128 ----
        v8f acc1[4] = {};
        #pragma unroll
        for (int kt = 0; kt < 2; ++kt) {        // K = 0..63 from feat16
            v16h b = ldrow16(fp + kt * 32 + hi * 16);
            #pragma unroll
            for (int mt = 0; mt < 4; ++mt)
                acc1[mt] = wmma_f16(ldA(lWn1, kt, mt, lane), b, acc1[mt]);
        }
        #pragma unroll
        for (int kt = 2; kt < 4; ++kt) {        // K = 64..127 from h_agg (f32)
            const float* cp = hp + (kt - 2) * 32 + hi * 16;
            v8f c0 = *(const v8f*)cp;
            v8f c1 = *(const v8f*)(cp + 8);
            v16h b;
            #pragma unroll
            for (int q = 0; q < 8; ++q) {
                b[q]     = (_Float16)c0[q];
                b[8 + q] = (_Float16)c1[q];
            }
            #pragma unroll
            for (int mt = 0; mt < 4; ++mt)
                acc1[mt] = wmma_f16(ldA(lWn1, kt, mt, lane), b, acc1[mt]);
        }
        #pragma unroll
        for (int mt = 0; mt < 4; ++mt) {
            acc1[mt] += *(const v8f*)&lbn1[mt * 16 + hi * 8];
            v8h hv;
            #pragma unroll
            for (int r = 0; r < 8; ++r) hv[r] = (_Float16)silu_f(acc1[mt][r]);
            *(v8h*)(trow + mt * 16 + hi * 8) = hv;
        }

        // ---- layer 2: Wn2^T @ (.) + bn2 ; vector global stores ----
        v8f acc2[4] = {};
        #pragma unroll
        for (int kt = 0; kt < 2; ++kt) {
            v16h b = ldrow16(trow + kt * 32 + hi * 16);
            #pragma unroll
            for (int mt = 0; mt < 4; ++mt)
                acc2[mt] = wmma_f16(ldA(lWn2, kt, mt, lane), b, acc2[mt]);
        }
        if (nv) {
            float* op = out_h + nd * 64 + hi * 8;
            #pragma unroll
            for (int mt = 0; mt < 4; ++mt) {
                v8f vo = acc2[mt] + *(const v8f*)&lbn2[mt * 16 + hi * 8];
                *(v8f*)(op + mt * 16) = vo;
            }
            // ---- x_new = coordinate + x_agg (low half only) ----
            if (!hi) {
                out_x[nd * 3 + 0] = coord[nd * 3 + 0] + x_agg[nd * 3 + 0];
                out_x[nd * 3 + 1] = coord[nd * 3 + 1] + x_agg[nd * 3 + 1];
                out_x[nd * 3 + 2] = coord[nd * 3 + 2] + x_agg[nd * 3 + 2];
            }
        }
    }
}

// ---------------------------------------------------------------------------
extern "C" void kernel_launch(void* const* d_in, const int* in_sizes, int n_in,
                              void* d_out, int out_size, void* d_ws, size_t ws_size,
                              hipStream_t stream) {
    const float* feat  = (const float*)d_in[0];
    const float* coord = (const float*)d_in[1];
    const int*   src   = (const int*)d_in[2];
    const int*   dst   = (const int*)d_in[3];
    const float* We1 = (const float*)d_in[4];  const float* be1 = (const float*)d_in[5];
    const float* We2 = (const float*)d_in[6];  const float* be2 = (const float*)d_in[7];
    const float* Wc1 = (const float*)d_in[8];  const float* bc1 = (const float*)d_in[9];
    const float* Wc2 = (const float*)d_in[10]; const float* bc2 = (const float*)d_in[11];
    const float* Wn1 = (const float*)d_in[12]; const float* bn1 = (const float*)d_in[13];
    const float* Wn2 = (const float*)d_in[14]; const float* bn2 = (const float*)d_in[15];

    const int Nn = in_sizes[0] / 64;   // feat is [N, 64]
    const int E  = in_sizes[2];        // src is [E]

    // workspace: [h_agg Nn*64 f32][x_agg Nn*3 f32][pad][feat16 Nn*64 f16]
    float* h_agg = (float*)d_ws;
    float* x_agg = h_agg + (size_t)Nn * 64;
    size_t zoff  = (((size_t)Nn * 67) + 63) & ~(size_t)63;   // 256B-align feat16
    _Float16* feat16 = (_Float16*)((float*)d_ws + zoff);

    float* out_h = (float*)d_out;                 // [Nn, 64]
    float* out_x = out_h + (size_t)Nn * 64;       // [Nn, 3]

    // prep: zero aggregation buffers + convert feat to f16
    long nz = (long)Nn * 67;
    long nf = (long)Nn * 64;
    long np = nz > nf ? nz : nf;
    egnn_prep<<<(unsigned)((np + TPB - 1) / TPB), TPB, 0, stream>>>(feat, feat16,
                                                                   h_agg, nz, nf);

    // edge pipeline
    long etiles = ((long)E + 15) / 16;
    long eper   = (long)WAVES_PER_BLOCK * EDGE_TPW;
    unsigned eblocks = (unsigned)((etiles + eper - 1) / eper);
    egnn_edge<<<eblocks, TPB, 0, stream>>>(feat16, coord, src, dst,
                                           We1, be1, We2, be2,
                                           Wc1, bc1, Wc2, bc2,
                                           h_agg, x_agg, E);

    // node pipeline
    long ntiles = ((long)Nn + 15) / 16;
    long nper   = (long)WAVES_PER_BLOCK * NODE_TPW;
    unsigned nblocks = (unsigned)((ntiles + nper - 1) / nper);
    egnn_node<<<nblocks, TPB, 0, stream>>>(feat16, coord,
                                           Wn1, bn1, Wn2, bn2,
                                           h_agg, x_agg,
                                           out_h, out_x, Nn);
}
